// LocalSelfAttentionBlock_87849261072775
// MI455X (gfx1250) — compile-verified, matrix operand order checked by reading, not run on hardware
//
#include <hip/hip_runtime.h>

// ---------------------------------------------------------------------------
// Local windowed self-attention block for MI455X (gfx1250, wave32, WMMA).
// B=2 S=4096 D=1024 H=8 W=1024 DH=128.  All matmuls via v_wmma_f32_16x16x32_bf16.
// GEMMs are register-blocked 64x64 per wave (16 WMMA / 8 fragment-loads per
// K-step -> 4x operand reuse).  Attention stages K/V^T tiles into LDS with
// gfx1250 async global->LDS copies (ASYNCcnt) instead of the VGPR round-trip.
// Working set (~125MB) fits the 192MB L2; HBM-minimal traffic ~100MB (~4.5us
// @23.3TB/s), so the implementation targets the matrix pipes, not VALU.
// ---------------------------------------------------------------------------

#define BB 2
#define SS 4096
#define DD 1024
#define HH 8
#define WW 1024
#define DH 128
#define NB 4               // S / W
#define MROWS (BB * SS)    // 8192

typedef __attribute__((ext_vector_type(16))) __bf16 bf16x16;
typedef __attribute__((ext_vector_type(8)))  float  f32x8;
typedef __attribute__((ext_vector_type(4)))  unsigned int u32x4;

__device__ __forceinline__ f32x8 zero8() {
  f32x8 z = {0.f, 0.f, 0.f, 0.f, 0.f, 0.f, 0.f, 0.f};
  return z;
}

// Load one 16x32 bf16 WMMA fragment for this lane.
// p points at (row-for-this-lane, k-chunk base); ISA 16-bit A layout:
//   elements 0..7  = K (8*half + 0..7), elements 8..15 = K (16 + 8*half + 0..7)
// Both halves are contiguous 16-byte runs -> two b128 loads.
__device__ __forceinline__ bf16x16 load_frag(const __bf16* p, int half) {
  union { u32x4 u[2]; bf16x16 v; } f;
  f.u[0] = *(const u32x4*)(p + half * 8);
  f.u[1] = *(const u32x4*)(p + 16 + half * 8);
  return f.v;
}

// gfx1250 async copy: 16B memory -> LDS, tracked by ASYNCcnt (no VGPR data).
// saddr form: vdst = LDS byte address, vaddr = 32-bit byte offset, saddr = base.
__device__ __forceinline__ void async_copy_b128(const __bf16* gbase, unsigned goff,
                                                __bf16* lds_ptr) {
  unsigned ldst = (unsigned)(unsigned long long)(uintptr_t)lds_ptr;
  unsigned long long sbase = (unsigned long long)(uintptr_t)gbase;
  asm volatile("global_load_async_to_lds_b128 %0, %1, %2"
               :: "v"(ldst), "v"(goff), "s"(sbase) : "memory");
}

__device__ __forceinline__ void wait_async_and_barrier() {
  asm volatile("s_wait_asynccnt 0x0" ::: "memory");
  __syncthreads();
}

// ---------------------------- conversion kernels ---------------------------

__global__ __launch_bounds__(256) void cvt_bf16_kernel(const float* __restrict__ src,
                                                       __bf16* __restrict__ dst, int n) {
  int i = blockIdx.x * 256 + threadIdx.x;
  if (i < n) dst[i] = (__bf16)src[i];
}

// w[K][N] fp32  ->  wT[N][K] bf16   (so B fragments load contiguous K runs)
__global__ __launch_bounds__(256) void cvt_transpose_kernel(const float* __restrict__ w,
                                                            __bf16* __restrict__ wt,
                                                            int K, int N) {
  int i = blockIdx.x * 256 + threadIdx.x;
  if (i < K * N) {
    int k = i / N, n = i % N;
    wt[(size_t)n * K + k] = (__bf16)w[i];
  }
}

// ------------------------------- QKV GEMM ----------------------------------
// C[8192, 3072] = val_bf @ w_qkvT^T + b_qkv.  64x64 tile per wave: 4 A-frags x
// 4 B-frags -> 16 WMMAs per 32-deep K step.  Scattered into q,k (row-major
// [bnh][w][dh]) and vT ([bnh][dh][w]) bf16 workspaces.

__global__ __launch_bounds__(256) void qkv_gemm_kernel(const __bf16* __restrict__ A,
                                                       const __bf16* __restrict__ Bt,
                                                       const float* __restrict__ bias,
                                                       __bf16* __restrict__ q_ws,
                                                       __bf16* __restrict__ k_ws,
                                                       __bf16* __restrict__ vt_ws) {
  const int wid  = blockIdx.x * 8 + (threadIdx.x >> 5);
  const int lane = threadIdx.x & 31;
  const int ln   = lane & 15, half = lane >> 4;
  const int tn = wid % (3 * DD / 64);   // 0..47
  const int tm = wid / (3 * DD / 64);   // 0..127

  const __bf16* arow[4];
  const __bf16* brow[4];
#pragma unroll
  for (int i = 0; i < 4; ++i) {
    arow[i] = A  + (size_t)(tm * 64 + i * 16 + ln) * DD;
    brow[i] = Bt + (size_t)(tn * 64 + i * 16 + ln) * DD;
  }

  f32x8 acc[4][4];
#pragma unroll
  for (int i = 0; i < 4; ++i)
#pragma unroll
    for (int j = 0; j < 4; ++j) acc[i][j] = zero8();

#pragma unroll 2
  for (int kk = 0; kk < DD; kk += 32) {
    bf16x16 af[4], bf[4];
#pragma unroll
    for (int i = 0; i < 4; ++i) af[i] = load_frag(arow[i] + kk, half);
#pragma unroll
    for (int j = 0; j < 4; ++j) bf[j] = load_frag(brow[j] + kk, half);
#pragma unroll
    for (int i = 0; i < 4; ++i)
#pragma unroll
      for (int j = 0; j < 4; ++j)
        acc[i][j] = __builtin_amdgcn_wmma_f32_16x16x32_bf16(false, af[i], false, bf[j],
                                                            (short)0, acc[i][j],
                                                            false, false);
  }

#pragma unroll
  for (int j = 0; j < 4; ++j) {
    const int col   = tn * 64 + j * 16 + ln;
    const float bv  = bias[col];
    const int which = col / DD;        // 0=q 1=k 2=v
    const int dcol  = col % DD;
    const int h     = dcol / DH, dh = dcol % DH;
#pragma unroll
    for (int i = 0; i < 4; ++i) {
#pragma unroll
      for (int r = 0; r < 8; ++r) {
        const int row  = tm * 64 + i * 16 + r + 8 * half;  // C-layout: M = r + 8*half
        const int b_   = row / SS;
        const int s    = row % SS;
        const int nwin = s / WW, w = s % WW;
        const size_t bnh = (size_t)((b_ * NB + nwin) * HH + h);
        const float v = acc[i][j][r] + bv;
        if (which == 0)      q_ws [bnh * WW * DH + (size_t)w * DH + dh] = (__bf16)v;
        else if (which == 1) k_ws [bnh * WW * DH + (size_t)w * DH + dh] = (__bf16)v;
        else                 vt_ws[bnh * DH * WW + (size_t)dh * WW + w] = (__bf16)v;
      }
    }
  }
}

// ------------------------------ attention ----------------------------------
// Flash-style over 1024 keys. Wave = 16 queries. We compute S^T = K·Q^T so the
// P^T accumulator C-layout (lane=query, VGPR=key) IS the fragment layout needed
// for ctx^T = V^T · P^T -- no transpose through LDS, per-lane softmax state.
// K/V^T tiles are staged into LDS with gfx1250 async global->LDS copies.

__global__ __launch_bounds__(256) void attn_kernel(const __bf16* __restrict__ q_ws,
                                                   const __bf16* __restrict__ k_ws,
                                                   const __bf16* __restrict__ vt_ws,
                                                   __bf16* __restrict__ ctx_ws) {
  __shared__ __bf16 ldsK[32 * DH];   // [key 0..31][dh 0..127]   8KB
  __shared__ __bf16 ldsV[DH * 32];   // [dh 0..127][key 0..31]   8KB

  const int bnh  = blockIdx.x >> 3;  // 0..63
  const int qblk = blockIdx.x & 7;   // 0..7
  const int wave = threadIdx.x >> 5;
  const int lane = threadIdx.x & 31;
  const int ln   = lane & 15, half = lane >> 4;

  const __bf16* qb = q_ws  + (size_t)bnh * WW * DH;
  const __bf16* kb = k_ws  + (size_t)bnh * WW * DH;
  const __bf16* vb = vt_ws + (size_t)bnh * DH * WW;

  const int qi = qblk * 128 + wave * 16 + ln;   // this lane's query (B-frag col)

  // Preload Q fragments (B operand of S^T WMMA): 4 chunks over DH=128
  bf16x16 qf[4];
#pragma unroll
  for (int t = 0; t < 4; ++t) qf[t] = load_frag(qb + (size_t)qi * DH + t * 32, half);

  f32x8 ctx[8];                       // ctx^T: tile t -> dh = 16t + r + 8*half
#pragma unroll
  for (int t = 0; t < 8; ++t) ctx[t] = zero8();
  float m_i = -3.0e38f, l_i = 0.f;

  for (int j = 0; j < 32; ++j) {      // 32-key chunks
    // async LDS fill: 512 x b128 each for K tile and V^T tile (no VGPR traffic)
    {
      const int tid = threadIdx.x;
#pragma unroll
      for (int it = 0; it < 2; ++it) {
        int idx = tid + it * 256;                 // 0..511
        int row = idx >> 4, c8 = (idx & 15) * 8;  // 16 x b128 per 128-wide row
        async_copy_b128(kb, (unsigned)(((j * 32 + row) * DH + c8) * 2),
                        ldsK + row * DH + c8);
      }
#pragma unroll
      for (int it = 0; it < 2; ++it) {
        int idx = tid + it * 256;
        int row = idx >> 2, c8 = (idx & 3) * 8;   // 4 x b128 per 32-wide row
        async_copy_b128(vb, (unsigned)((row * WW + j * 32 + c8) * 2),
                        ldsV + row * 32 + c8);
      }
    }
    wait_async_and_barrier();

    // S^T tiles: M=key(16), N=query(16), K=DH via 4 WMMAs each
    f32x8 st[2];
    st[0] = zero8(); st[1] = zero8();
#pragma unroll
    for (int t2 = 0; t2 < 2; ++t2) {
#pragma unroll
      for (int kk = 0; kk < 4; ++kk) {
        bf16x16 kf = load_frag(ldsK + (t2 * 16 + ln) * DH + kk * 32, half);
        st[t2] = __builtin_amdgcn_wmma_f32_16x16x32_bf16(false, kf, false, qf[kk],
                                                         (short)0, st[t2], false, false);
      }
    }

    // online softmax (row = query = lane%16; keys live in VGPRs + half pair)
    const float scale = 0.08838834764831845f;     // 1/sqrt(128)
    float mx = -3.0e38f;
#pragma unroll
    for (int t2 = 0; t2 < 2; ++t2)
#pragma unroll
      for (int r = 0; r < 8; ++r) { st[t2][r] *= scale; mx = fmaxf(mx, st[t2][r]); }
    mx = fmaxf(mx, __shfl_xor(mx, 16, 32));       // combine key halves
    const float nm    = fmaxf(m_i, mx);
    const float alpha = __expf(m_i - nm);
    float rs = 0.f;
#pragma unroll
    for (int t2 = 0; t2 < 2; ++t2)
#pragma unroll
      for (int r = 0; r < 8; ++r) { st[t2][r] = __expf(st[t2][r] - nm); rs += st[t2][r]; }
    rs += __shfl_xor(rs, 16, 32);
    l_i = l_i * alpha + rs;
    m_i = nm;
#pragma unroll
    for (int t = 0; t < 8; ++t)
#pragma unroll
      for (int r = 0; r < 8; ++r) ctx[t][r] *= alpha;   // per-lane rescale

    // Pack P^T: C-layout VGPR r = key (8*half + r) maps 1:1 onto the 16x32
    // fragment element order (pairs per VGPR, tile1 -> elements 8..15).
    bf16x16 pv;
#pragma unroll
    for (int e = 0; e < 8; ++e) pv[e] = (__bf16)st[0][e];
#pragma unroll
    for (int e = 0; e < 8; ++e) pv[8 + e] = (__bf16)st[1][e];

    // ctx^T += V^T(16dh x 32key) @ P^T(32key x 16query), 8 dh-tiles
#pragma unroll
    for (int t = 0; t < 8; ++t) {
      bf16x16 vf = load_frag(ldsV + (t * 16 + ln) * 32, half);
      ctx[t] = __builtin_amdgcn_wmma_f32_16x16x32_bf16(false, vf, false, pv, (short)0,
                                                       ctx[t], false, false);
    }
    __syncthreads();
  }

  // normalize + store ctx as bf16 [token][D] row-major for the out-proj GEMM
  const float inv_l = 1.0f / l_i;
  const int b    = bnh >> 5;
  const int nwin = (bnh >> 3) & 3;
  const int h    = bnh & 7;
  const size_t token = (size_t)b * SS + (size_t)nwin * WW + qi;
  __bf16* crow = ctx_ws + token * DD + h * DH;
#pragma unroll
  for (int t = 0; t < 8; ++t)
#pragma unroll
    for (int r = 0; r < 8; ++r)
      crow[t * 16 + 8 * half + r] = (__bf16)(ctx[t][r] * inv_l);
}

// ----------------------------- out-proj GEMM -------------------------------
// 64x64 tile per wave, same register blocking as the QKV GEMM.

__global__ __launch_bounds__(256) void out_gemm_kernel(const __bf16* __restrict__ A,
                                                       const __bf16* __restrict__ Bt,
                                                       const float* __restrict__ bias,
                                                       float* __restrict__ Cout) {
  const int wid  = blockIdx.x * 8 + (threadIdx.x >> 5);
  const int lane = threadIdx.x & 31;
  const int ln   = lane & 15, half = lane >> 4;
  const int tn = wid % (DD / 64);   // 0..15
  const int tm = wid / (DD / 64);   // 0..127

  const __bf16* arow[4];
  const __bf16* brow[4];
#pragma unroll
  for (int i = 0; i < 4; ++i) {
    arow[i] = A  + (size_t)(tm * 64 + i * 16 + ln) * DD;
    brow[i] = Bt + (size_t)(tn * 64 + i * 16 + ln) * DD;
  }

  f32x8 acc[4][4];
#pragma unroll
  for (int i = 0; i < 4; ++i)
#pragma unroll
    for (int j = 0; j < 4; ++j) acc[i][j] = zero8();

#pragma unroll 2
  for (int kk = 0; kk < DD; kk += 32) {
    bf16x16 af[4], bf[4];
#pragma unroll
    for (int i = 0; i < 4; ++i) af[i] = load_frag(arow[i] + kk, half);
#pragma unroll
    for (int j = 0; j < 4; ++j) bf[j] = load_frag(brow[j] + kk, half);
#pragma unroll
    for (int i = 0; i < 4; ++i)
#pragma unroll
      for (int j = 0; j < 4; ++j)
        acc[i][j] = __builtin_amdgcn_wmma_f32_16x16x32_bf16(false, af[i], false, bf[j],
                                                            (short)0, acc[i][j],
                                                            false, false);
  }

#pragma unroll
  for (int j = 0; j < 4; ++j) {
    const int col  = tn * 64 + j * 16 + ln;
    const float bv = bias[col];
#pragma unroll
    for (int i = 0; i < 4; ++i)
#pragma unroll
      for (int r = 0; r < 8; ++r) {
        const int row = tm * 64 + i * 16 + r + 8 * half;
        Cout[(size_t)row * DD + col] = acc[i][j][r] + bv;
      }
  }
}

// --------------------------- residual + layernorm --------------------------

__global__ __launch_bounds__(256) void ln_kernel(const float* __restrict__ val,
                                                 const float* __restrict__ attn,
                                                 const float* __restrict__ gamma,
                                                 const float* __restrict__ beta,
                                                 float* __restrict__ out) {
  const int row = blockIdx.x;
  const float* v = val  + (size_t)row * DD;
  const float* a = attn + (size_t)row * DD;

  float x[4], s = 0.f, s2 = 0.f;
#pragma unroll
  for (int i = 0; i < 4; ++i) {
    int c = threadIdx.x + i * 256;
    x[i] = v[c] + a[c];
    s += x[i]; s2 += x[i] * x[i];
  }
#pragma unroll
  for (int off = 16; off; off >>= 1) {
    s  += __shfl_xor(s, off, 32);
    s2 += __shfl_xor(s2, off, 32);
  }
  __shared__ float red[2][8];
  const int wave = threadIdx.x >> 5, lane = threadIdx.x & 31;
  if (lane == 0) { red[0][wave] = s; red[1][wave] = s2; }
  __syncthreads();
  if (threadIdx.x == 0) {
    float ts = 0.f, ts2 = 0.f;
    for (int i = 0; i < 8; ++i) { ts += red[0][i]; ts2 += red[1][i]; }
    red[0][0] = ts; red[1][0] = ts2;
  }
  __syncthreads();
  const float mu  = red[0][0] * (1.0f / DD);
  const float var = red[1][0] * (1.0f / DD) - mu * mu;
  const float inv = rsqrtf(var + 1e-5f);
#pragma unroll
  for (int i = 0; i < 4; ++i) {
    int c = threadIdx.x + i * 256;
    out[(size_t)row * DD + c] = (x[i] - mu) * inv * gamma[c] + beta[c];
  }
}

// -------------------------------- launcher ---------------------------------

extern "C" void kernel_launch(void* const* d_in, const int* in_sizes, int n_in,
                              void* d_out, int out_size, void* d_ws, size_t ws_size,
                              hipStream_t stream) {
  const float* val      = (const float*)d_in[0];
  const float* w_qkv    = (const float*)d_in[1];
  const float* b_qkv    = (const float*)d_in[2];
  const float* w_out    = (const float*)d_in[3];
  const float* b_out    = (const float*)d_in[4];
  const float* ln_gamma = (const float*)d_in[5];
  const float* ln_beta  = (const float*)d_in[6];
  float* out = (float*)d_out;

  // workspace layout (bytes); total ~125.8 MB
  char* ws = (char*)d_ws;
  size_t off = 0;
  __bf16* val_bf = (__bf16*)(ws + off); off += (size_t)MROWS * DD * 2;       // 16MB
  __bf16* wqkvT  = (__bf16*)(ws + off); off += (size_t)3 * DD * DD * 2;      //  6MB
  __bf16* woutT  = (__bf16*)(ws + off); off += (size_t)DD * DD * 2;          //  2MB
  __bf16* q_ws   = (__bf16*)(ws + off); off += (size_t)MROWS * DD * 2;       // 16MB
  __bf16* k_ws   = (__bf16*)(ws + off); off += (size_t)MROWS * DD * 2;       // 16MB
  __bf16* vt_ws  = (__bf16*)(ws + off); off += (size_t)MROWS * DD * 2;       // 16MB
  __bf16* ctx_ws = (__bf16*)(ws + off); off += (size_t)MROWS * DD * 2;       // 16MB
  float*  aout   = (float*) (ws + off); off += (size_t)MROWS * DD * 4;       // 32MB
  (void)ws_size; (void)in_sizes; (void)n_in; (void)out_size;

  // 1. fp32 -> bf16 conversions / weight transposes
  {
    int n = MROWS * DD;
    cvt_bf16_kernel<<<(n + 255) / 256, 256, 0, stream>>>(val, val_bf, n);
    cvt_transpose_kernel<<<(3 * DD * DD + 255) / 256, 256, 0, stream>>>(w_qkv, wqkvT, DD, 3 * DD);
    cvt_transpose_kernel<<<(DD * DD + 255) / 256, 256, 0, stream>>>(w_out, woutT, DD, DD);
  }
  // 2. QKV projection, 64x64 tiles: (8192/64)*(3072/64)=6144 waves / 8
  qkv_gemm_kernel<<<(MROWS / 64) * (3 * DD / 64) / 8, 256, 0, stream>>>(
      val_bf, wqkvT, b_qkv, q_ws, k_ws, vt_ws);
  // 3. windowed flash attention: 64 (b,win,head) x 8 query-blocks
  attn_kernel<<<64 * 8, 256, 0, stream>>>(q_ws, k_ws, vt_ws, ctx_ws);
  // 4. output projection, 64x64 tiles: (8192/64)*(1024/64)=2048 waves / 8
  out_gemm_kernel<<<(MROWS / 64) * (DD / 64) / 8, 256, 0, stream>>>(
      ctx_ws, woutT, b_out, aout);
  // 5. residual + layernorm
  ln_kernel<<<MROWS, 256, 0, stream>>>(val, aout, ln_gamma, ln_beta, out);
}
